// EdgeConv_85779086836263
// MI455X (gfx1250) — matrix operand order repaired and checked by reading, not compiled, not run on hardware
//
#include <hip/hip_runtime.h>
#include <hip/hip_bf16.h>
#include <math.h>

// Problem constants (match reference)
#define B_   16
#define C_   64
#define N_   2048
#define K_   20
#define OUT_ 64
#define C2_  128
#define EPS_ 1e-5f
#define NEG_SLOPE_ 0.2f

typedef __attribute__((ext_vector_type(16))) _Float16 v16h;
typedef __attribute__((ext_vector_type(8)))  _Float16 v8h;
typedef __attribute__((ext_vector_type(8)))  float    v8f;

#define WMMA_F16(A, Bm, Cm) \
  __builtin_amdgcn_wmma_f32_16x16x32_f16(false, (A), false, (Bm), (short)0, (Cm), false, false)

// ---------------------------------------------------------------------------
// Workspace layout (bytes), all offsets 256-aligned
//   xt   : f16 TRANSPOSED x, [B][N][C]         @ 0         (4,194,304)
//   idx  : int  neighbor idx, [B][N][K]        @ 4,194,304 (2,621,440)
//   wc   : f16  central weights  [OUT][C]      @ 6,815,744 (8,192)
//   wn   : f16  neighbor weights [K][OUT][C]   @ 6,823,936 (163,840)
//   sq   : f32  squared norms [B][N]           @ 6,987,776 (131,072)
//   stat : f32  per-channel {mean, invstd}[64] @ 7,118,848 (512)
// ---------------------------------------------------------------------------
#define WS_XT   0
#define WS_IDX  4194304
#define WS_WC   6815744
#define WS_WN   6823936
#define WS_SQ   6987776
#define WS_STAT 7118848

// ---- CDNA5 async copy: global -> LDS, 16 bytes, tracked by ASYNCcnt --------
__device__ __forceinline__ void async_copy_b128(uint32_t lds_addr, const void* gaddr) {
  asm volatile("global_load_async_to_lds_b128 %0, %1, off"
               :: "v"(lds_addr), "v"(gaddr) : "memory");
}

// ---- WMMA fragment loaders (all sources row-major with row stride 64) ------
// A-matrix (16x32 f16, MxK): M = lane&15; halves 0..15 -> K = klo+hi16+h,
// hi16 = 16 for lanes 16-31.
// One contiguous 16-half run -> two b128 loads + shuffle.
__device__ __forceinline__ v16h load_afrag64(const _Float16* m, int row, int klo,
                                             int lane) {
  const int hi8 = (lane & 16) ? 16 : 0;
  const _Float16* r = m + (size_t)row * 64 + klo + hi8;
  const v8h lo = *(const v8h*)r;
  const v8h hh = *(const v8h*)(r + 8);
  return __builtin_shufflevector(lo, hh, 0, 1, 2, 3, 4, 5, 6, 7,
                                 8, 9, 10, 11, 12, 13, 14, 15);
}

// B-matrix (32x16 f16, KxN): column n = lane&15;
// halves 0..15 -> K = klo + (lane>=16 ? 16 : 0) + h  -> one contiguous 32B run.
__device__ __forceinline__ v16h load_bfrag64(const _Float16* m, int point, int klo,
                                             int lane) {
  const int koff = (lane & 16) ? 16 : 0;
  return *(const v16h*)(m + (size_t)point * 64 + klo + koff);
}

// ---- Kernel 0: f32 [B][C][N] -> f16 TRANSPOSED [B][N][C] -------------------
__global__ void cvt_kernel(const float* __restrict__ x, _Float16* __restrict__ xt) {
  const int i = blockIdx.x * blockDim.x + threadIdx.x;  // = b*C*N + c*N + n
  if (i >= B_ * C_ * N_) return;
  const int n  = i & (N_ - 1);
  const int bc = i >> 11;            // N_ = 2^11
  const int c  = bc & (C_ - 1);
  const int b  = bc >> 6;            // C_ = 2^6
  xt[((size_t)b * N_ + n) * C_ + c] = (_Float16)x[i];
}

// ---- Kernel 1: squared norms sq[b][n] = sum_c x^2 (f32, exact) -------------
__global__ void sq_kernel(const float* __restrict__ x, float* __restrict__ sq) {
  const int i = blockIdx.x * blockDim.x + threadIdx.x;
  if (i >= B_ * N_) return;
  const int b = i / N_, n = i - b * N_;
  const float* xc = x + (size_t)b * C_ * N_ + n;
  float s = 0.f;
#pragma unroll 4
  for (int c = 0; c < C_; ++c) {
    const float v = xc[(size_t)c * N_];
    s += v * v;
  }
  sq[i] = s;
}

// ---- Kernel 2: weight transforms (central-combined + per-neighbor-slot) ----
__global__ void wprep_kernel(const float* __restrict__ w,
                             _Float16* __restrict__ wc,
                             _Float16* __restrict__ wn) {
  const int t = blockIdx.x * blockDim.x + threadIdx.x;
  if (t >= OUT_ * C_) return;
  const int o = t / C_, c = t - o * C_;
  const float* wo = w + (size_t)o * C2_ * K_;
  float s = 0.f;
#pragma unroll
  for (int j = 0; j < K_; ++j) {
    const float wctr = wo[(size_t)c * K_ + j];
    const float wnbr = wo[(size_t)(C_ + c) * K_ + j];
    s += wctr - wnbr;
    wn[((size_t)j * OUT_ + o) * C_ + c] = (_Float16)wnbr;
  }
  wc[(size_t)o * C_ + c] = (_Float16)s;
}

// ---- Kernel 3: fused Gram-WMMA distance + top-(K+1) selection --------------
// One workgroup = one batch, one 16-row i-tile. 4 waves split the 2048 j-cols.
// Full 16x2048 f32 distance block lives in LDS (128 KB; CDNA5 WGP = 320 KB).
__global__ void knn_kernel(const _Float16* __restrict__ xt,
                           const float* __restrict__ sq,
                           int* __restrict__ idxbuf) {
  extern __shared__ float distbuf[];  // 16 * 2048 floats
  const int b    = blockIdx.y;
  const int i0   = blockIdx.x * 16;
  const int lane = threadIdx.x & 31;
  const int wave = threadIdx.x >> 5;
  const int lm   = lane & 15;
  const int hi   = (lane >> 4) & 1;
  const _Float16* xtb = xt + (size_t)b * N_ * C_;
  const float*    sqb = sq + (size_t)b * N_;

  // A fragments for this i-tile (reused across all 32 j-tiles per wave)
  const v16h a0 = load_afrag64(xtb, i0 + lm, 0,  lane);
  const v16h a1 = load_afrag64(xtb, i0 + lm, 32, lane);

  float sqi[8];
#pragma unroll
  for (int g = 0; g < 8; ++g) sqi[g] = sqb[i0 + g + hi * 8];

  // dist = |xi|^2 + |xj|^2 - 2 xi.xj  (dot via WMMA, K=64 in two k-steps)
  for (int jt = wave; jt < N_ / 16; jt += 4) {
    const int j0 = jt * 16;
    const v16h b0 = load_bfrag64(xtb, j0 + lm, 0,  lane);
    const v16h b1 = load_bfrag64(xtb, j0 + lm, 32, lane);
    v8f acc = {};
    acc = WMMA_F16(a0, b0, acc);
    acc = WMMA_F16(a1, b1, acc);
    const int   j   = j0 + lm;
    const float sqj = sqb[j];
#pragma unroll
    for (int g = 0; g < 8; ++g)
      distbuf[(size_t)(g + hi * 8) * N_ + j] = sqi[g] + sqj - 2.0f * acc[g];
  }
  __syncthreads();

  // Iterative min-extraction: 21 mins per row; first one (self) is dropped.
  for (int r = wave; r < 16; r += 4) {
    const int gi = i0 + r;
    volatile float* row = distbuf + (size_t)r * N_;
    for (int sel = 0; sel <= K_; ++sel) {
      float best = 3.0e38f;
      int   bidx = N_;
      for (int cj = lane; cj < N_; cj += 32) {
        const float v = row[cj];
        if (v < best) { best = v; bidx = cj; }
      }
#pragma unroll
      for (int off = 16; off; off >>= 1) {  // wave32 butterfly, all lanes get min
        const float ov = __shfl_xor(best, off);
        const int   oi = __shfl_xor(bidx, off);
        if (ov < best || (ov == best && oi < bidx)) { best = ov; bidx = oi; }
      }
      if (sel > 0 && lane == 0)
        idxbuf[((size_t)b * N_ + gi) * K_ + (sel - 1)] = bidx;
      if (lane == 0) row[bidx] = 3.0e38f;  // DS ops in-order per wave
    }
  }
}

// ---- Kernel 4: edge-conv GEMM with async-LDS double-buffered weights -------
// Workgroup = one batch, one 16-col n-tile. Wave w owns output rows o0=16w.
// wn[j] tiles (8 KB) stream into LDS via global_load_async_to_lds_b128;
// thread t copies bytes [64t, 64t+64) => wave w stages exactly its own rows
// 16w..16w+15, so only a per-wave s_wait_asynccnt is needed (no barrier).
__global__ void edgeconv_kernel(const _Float16* __restrict__ xt,
                                const _Float16* __restrict__ wc,
                                const _Float16* __restrict__ wn,
                                const int* __restrict__ idxbuf,
                                const float* __restrict__ bias,
                                float* __restrict__ y) {
  __shared__ __align__(16) _Float16 wbuf[2][OUT_ * C_];  // 2 x 8 KB
  const int b    = blockIdx.y;
  const int n0   = blockIdx.x * 16;
  const int tid  = threadIdx.x;
  const int lane = tid & 31;
  const int wave = tid >> 5;
  const int lm   = lane & 15;
  const int hi   = (lane >> 4) & 1;
  const int o0   = wave * 16;
  const _Float16* xtb = xt + (size_t)b * N_ * C_;
  const int n = n0 + lm;

  const char*    wng  = (const char*)wn;
  const uint32_t lds0 = (uint32_t)(uintptr_t)(&wbuf[0][0]);
  const uint32_t lds1 = (uint32_t)(uintptr_t)(&wbuf[1][0]);
  const int      toff = tid * 64;  // 128 threads x 64 B = one 8 KB tile

  // Prologue: stage wn[0] into wbuf[0]
#pragma unroll
  for (int k = 0; k < 4; ++k)
    async_copy_b128(lds0 + toff + k * 16, wng + toff + k * 16);

  v8f acc = {};
  {  // central term: W_c (64x64) x X[:, n-tile]  (overlaps the async stage)
    const v16h wa0 = load_afrag64(wc, o0 + lm, 0,  lane);
    const v16h wa1 = load_afrag64(wc, o0 + lm, 32, lane);
    const v16h xb0 = load_bfrag64(xtb, n, 0,  lane);
    const v16h xb1 = load_bfrag64(xtb, n, 32, lane);
    acc = WMMA_F16(wa0, xb0, acc);
    acc = WMMA_F16(wa1, xb1, acc);
  }

  // gathered neighbor terms: W_n^j x X[:, idx[n, j]]
  const int* ib = idxbuf + ((size_t)b * N_ + n) * K_;
  for (int j = 0; j < K_; ++j) {
    if (j + 1 < K_) {
      const uint32_t dst = ((j + 1) & 1) ? lds1 : lds0;
      const char*    src = wng + (size_t)(j + 1) * (OUT_ * C_ * 2);
#pragma unroll
      for (int k = 0; k < 4; ++k)
        async_copy_b128(dst + toff + k * 16, src + toff + k * 16);
      // 8 outstanding -> wait until only the 4 just-issued remain
      asm volatile("s_wait_asynccnt 4" ::: "memory");
    } else {
      asm volatile("s_wait_asynccnt 0" ::: "memory");
    }
    const _Float16* wj = (j & 1) ? &wbuf[1][0] : &wbuf[0][0];
    const int nb = ib[j];  // per-lane gather column (contiguous 64 ch in xt)
    const v16h wa0 = load_afrag64(wj, o0 + lm, 0,  lane);
    const v16h wa1 = load_afrag64(wj, o0 + lm, 32, lane);
    const v16h xb0 = load_bfrag64(xtb, nb, 0,  lane);
    const v16h xb1 = load_bfrag64(xtb, nb, 32, lane);
    acc = WMMA_F16(wa0, xb0, acc);
    acc = WMMA_F16(wa1, xb1, acc);
  }

#pragma unroll
  for (int g = 0; g < 8; ++g) {
    const int o = o0 + g + hi * 8;
    y[((size_t)b * OUT_ + o) * N_ + n] = acc[g] + bias[o];
  }
}

// ---- Kernel 5: per-channel mean / inv-std over (B, N) ----------------------
__global__ void stats_kernel(const float* __restrict__ y, float* __restrict__ stats) {
  __shared__ float s1[256], s2[256];
  const int o = blockIdx.x;
  float a = 0.f, q = 0.f;
  for (int i = threadIdx.x; i < B_ * N_; i += 256) {
    const int bb = i / N_, n = i - bb * N_;
    const float v = y[((size_t)bb * OUT_ + o) * N_ + n];
    a += v; q += v * v;
  }
  s1[threadIdx.x] = a; s2[threadIdx.x] = q;
  __syncthreads();
  for (int s = 128; s > 0; s >>= 1) {
    if (threadIdx.x < s) {
      s1[threadIdx.x] += s1[threadIdx.x + s];
      s2[threadIdx.x] += s2[threadIdx.x + s];
    }
    __syncthreads();
  }
  if (threadIdx.x == 0) {
    const float inv  = 1.0f / (float)(B_ * N_);
    const float mean = s1[0] * inv;
    const float var  = s2[0] * inv - mean * mean;
    stats[o * 2]     = mean;
    stats[o * 2 + 1] = rsqrtf(var + EPS_);
  }
}

// ---- Kernel 6: normalize + leaky ReLU (in place on d_out) ------------------
__global__ void norm_kernel(float* __restrict__ y,
                            const float* __restrict__ stats,
                            const float* __restrict__ gamma,
                            const float* __restrict__ beta) {
  const int total = B_ * OUT_ * N_;
  const int i = blockIdx.x * blockDim.x + threadIdx.x;
  if (i >= total) return;
  const int o = (i / N_) & (OUT_ - 1);
  float v = (y[i] - stats[o * 2]) * stats[o * 2 + 1] * gamma[o] + beta[o];
  y[i] = v >= 0.f ? v : NEG_SLOPE_ * v;
}

// ---------------------------------------------------------------------------
extern "C" void kernel_launch(void* const* d_in, const int* in_sizes, int n_in,
                              void* d_out, int out_size, void* d_ws, size_t ws_size,
                              hipStream_t stream) {
  (void)in_sizes; (void)n_in; (void)out_size; (void)ws_size;
  const float* x     = (const float*)d_in[0];
  const float* w     = (const float*)d_in[1];
  const float* bias  = (const float*)d_in[2];
  const float* gamma = (const float*)d_in[3];
  const float* beta  = (const float*)d_in[4];
  float* y = (float*)d_out;

  char* ws = (char*)d_ws;
  _Float16* xt   = (_Float16*)(ws + WS_XT);
  int*      idxb = (int*)     (ws + WS_IDX);
  _Float16* wc   = (_Float16*)(ws + WS_WC);
  _Float16* wn   = (_Float16*)(ws + WS_WN);
  float*    sq   = (float*)   (ws + WS_SQ);
  float*    st   = (float*)   (ws + WS_STAT);

  // 0) x -> f16, transposed to [B][N][C] so all WMMA fragments are contiguous
  cvt_kernel<<<dim3((B_ * C_ * N_) / 256), dim3(256), 0, stream>>>(x, xt);
  // 1) squared norms (exact f32)
  sq_kernel<<<dim3((B_ * N_) / 256), dim3(256), 0, stream>>>(x, sq);
  // 2) weight transforms
  wprep_kernel<<<dim3((OUT_ * C_) / 256), dim3(256), 0, stream>>>(w, wc, wn);
  // 3) fused WMMA distances + top-k (128 KB dynamic LDS per WG)
  knn_kernel<<<dim3(N_ / 16, B_), dim3(128), 16 * N_ * sizeof(float), stream>>>(
      xt, sq, idxb);
  // 4) edge-conv WMMA GEMMs (async-LDS weight streaming + gathered B)
  edgeconv_kernel<<<dim3(N_ / 16, B_), dim3(128), 0, stream>>>(
      xt, wc, wn, idxb, bias, y);
  // 5) batch-norm statistics
  stats_kernel<<<dim3(OUT_), dim3(256), 0, stream>>>(y, st);
  // 6) normalize + leaky ReLU
  norm_kernel<<<dim3((B_ * OUT_ * N_ + 255) / 256), dim3(256), 0, stream>>>(
      y, st, gamma, beta);
}
